// CausalSelfAttention_84731114815967
// MI455X (gfx1250) — matrix operand order repaired and checked by reading, not compiled, not run on hardware
//
#include <hip/hip_runtime.h>

// ---------------------------------------------------------------------------
// Causal self-attention with RoPE for MI455X (gfx1250, wave32, WMMA).
// GEMMs: v_wmma_f32_16x16x32_bf16, 4 accumulators per wave (16x64 tile) to
// amortize A fragments and LDS staging; V tiles staged via
// global_load_async_to_lds_b128 (ASYNCcnt); softmax reductions via DPP16.
// ---------------------------------------------------------------------------

typedef __attribute__((ext_vector_type(16))) __bf16        v16bf;
typedef __attribute__((ext_vector_type(8)))  float         v8f;
typedef __attribute__((ext_vector_type(4)))  unsigned int  v4u;

union Frag16 {
    v16bf v;
    unsigned short u[16];
    v4u q2[2];
};
union RowBuf {
    v4u q[4];
    unsigned short u[32];
};

static constexpr int BATCH = 2;
static constexpr int SEQ   = 2048;
static constexpr int DM    = 1024;
static constexpr int NH    = 16;
static constexpr int DK    = 64;
static constexpr int BS    = BATCH * SEQ;      // 4096 rows

__device__ __forceinline__ unsigned short f2bf(float f) {
    unsigned int u = __float_as_uint(f);
    unsigned int r = u + 0x7FFFu + ((u >> 16) & 1u);   // round-to-nearest-even
    return (unsigned short)(r >> 16);
}
__device__ __forceinline__ float bf2f(unsigned short h) {
    return __uint_as_float(((unsigned int)h) << 16);
}

// ---- DPP16 cross-lane helpers: butterfly reduction within each 16-lane row.
template <int CTRL>
__device__ __forceinline__ float dpp_mv(float x) {
    int xi = __float_as_int(x);
    int yi = __builtin_amdgcn_update_dpp(xi, xi, CTRL, 0xF, 0xF, true);
    return __int_as_float(yi);
}
__device__ __forceinline__ float row16_max(float x) {
    x = fmaxf(x, dpp_mv<0xB1>(x));    // quad_perm(1,0,3,2)  = xor1
    x = fmaxf(x, dpp_mv<0x4E>(x));    // quad_perm(2,3,0,1)  = xor2
    x = fmaxf(x, dpp_mv<0x141>(x));   // row_half_mirror     = xor7
    x = fmaxf(x, dpp_mv<0x140>(x));   // row_mirror          = xor15
    return x;
}
__device__ __forceinline__ float row16_sum(float x) {
    x += dpp_mv<0xB1>(x);
    x += dpp_mv<0x4E>(x);
    x += dpp_mv<0x141>(x);
    x += dpp_mv<0x140>(x);
    return x;
}

// ---------------------------------------------------------------------------
// Kernel 1: fp32 -> bf16 conversion (2 elems/thread, packed b32 stores)
// ---------------------------------------------------------------------------
__global__ void cvt_f32_bf16(const float* __restrict__ src,
                             unsigned int* __restrict__ dst, int npairs) {
    int i = blockIdx.x * blockDim.x + threadIdx.x;
    if (i < npairs) {
        float2 f = reinterpret_cast<const float2*>(src)[i];
        dst[i] = (unsigned)f2bf(f.x) | ((unsigned)f2bf(f.y) << 16);
    }
}

// ---------------------------------------------------------------------------
// Kernel 2/5: C[M,N] = A[M,K] * W[N,K]^T  (bf16 in, fp32 acc, bf16/f32 out)
// Block: 256 threads = 8 waves stacked along M (block tile 128 x 64).
// Each wave owns a 16x64 output strip: one A fragment drives 4 WMMAs per
// 32-wide k chunk. W tile (32k x 64n, transposed) staged once per block in
// LDS; B fragments are 16 strided 16-bit reads (stride 64).
// ---------------------------------------------------------------------------
template <bool F32OUT>
__global__ void gemm_xwT(const unsigned short* __restrict__ A,
                         const unsigned short* __restrict__ W,
                         void* __restrict__ Cout, int M, int N, int K) {
    __shared__ unsigned short WldsT[32 * 64];        // [k][n], 4 KB

    const int lane = threadIdx.x & 31;
    const int wave = threadIdx.x >> 5;
    const int n0   = blockIdx.x * 64;
    const int m0   = blockIdx.y * 128 + wave * 16;
    const int hi   = lane >> 4;
    const int l15  = lane & 15;
    const int kb   = l15 + 16 * hi;                  // B-fragment K per lane

    const unsigned short* arow = A + (size_t)(m0 + l15) * K;

    v8f acc[4] = {};
    for (int k0 = 0; k0 < K; k0 += 32) {
        // cooperative transposed W-tile store: 2048 elems / 256 threads = 8
        {
#pragma unroll
            for (int r = 0; r < 2; ++r) {
                int idx = threadIdx.x * 8 + r * 4;   // (n,k) flat: n*32 + kk
                int n = idx >> 5, kk = idx & 31;
                const unsigned short* wp = W + (size_t)(n0 + n) * K + k0 + kk;
                uint2 two4 = *reinterpret_cast<const uint2*>(wp);
                WldsT[(kk + 0) * 64 + n] = (unsigned short)(two4.x & 0xFFFFu);
                WldsT[(kk + 1) * 64 + n] = (unsigned short)(two4.x >> 16);
                WldsT[(kk + 2) * 64 + n] = (unsigned short)(two4.y & 0xFFFFu);
                WldsT[(kk + 3) * 64 + n] = (unsigned short)(two4.y >> 16);
                if (k0 + 32 < K) __builtin_prefetch(wp + 32, 0, 1);
            }
        }
        __syncthreads();

        // one A fragment per chunk: two contiguous 8-element runs per lane
        Frag16 a;
        a.q2[0] = *reinterpret_cast<const v4u*>(arow + k0 + 8 * hi);
        a.q2[1] = *reinterpret_cast<const v4u*>(arow + k0 + 16 + 8 * hi);

        // 4 B fragments / 4 WMMAs: column 16t+l15 of [k][n] tile, k in table order
#pragma unroll
        for (int t = 0; t < 4; ++t) {
            Frag16 bfr;
#pragma unroll
            for (int s = 0; s < 16; ++s)
                bfr.u[s] = WldsT[(8 * hi + s + (s & 8)) * 64 + 16 * t + l15];
            acc[t] = __builtin_amdgcn_wmma_f32_16x16x32_bf16(
                false, a.v, false, bfr.v, (short)0, acc[t], false, false);
        }
        __syncthreads();
    }

    // C layout: element (m = v + 8*hi, n = 16*t + l15)
#pragma unroll
    for (int t = 0; t < 4; ++t)
#pragma unroll
        for (int v = 0; v < 8; ++v) {
            int m = v + 8 * hi;
            size_t off = (size_t)(m0 + m) * N + (n0 + 16 * t + l15);
            if constexpr (F32OUT)
                reinterpret_cast<float*>(Cout)[off] = acc[t][v];
            else
                reinterpret_cast<unsigned short*>(Cout)[off] = f2bf(acc[t][v]);
        }
}

// ---------------------------------------------------------------------------
// Kernel 3: RoPE in-place on Q and K (bf16 storage, fp32 math)
// ---------------------------------------------------------------------------
__global__ void rope_kernel(unsigned short* __restrict__ q,
                            unsigned short* __restrict__ k,
                            const int* __restrict__ pos) {
    int idx = blockIdx.x * blockDim.x + threadIdx.x;   // pair index
    const int pairsPerRow = DM / 2;
    if (idx >= BS * pairsPerRow) return;
    int row = idx / pairsPerRow;
    int col = (idx % pairsPerRow) * 2;
    int i   = (col & (DK - 1)) >> 1;                   // 0..31 within head

    float p   = (float)pos[row];
    float inv = __expf(-((float)(2 * i) / (float)DK) * 9.210340371976184f);
    float ang = p * inv;
    float s, c;
    __sincosf(ang, &s, &c);

    size_t o = (size_t)row * DM + col;
    float x1 = bf2f(q[o]), x2 = bf2f(q[o + 1]);
    q[o]     = f2bf(x1 * c - x2 * s);
    q[o + 1] = f2bf(x2 * c + x1 * s);
    x1 = bf2f(k[o]); x2 = bf2f(k[o + 1]);
    k[o]     = f2bf(x1 * c - x2 * s);
    k[o + 1] = f2bf(x2 * c + x1 * s);
}

// ---------------------------------------------------------------------------
// Kernel 4: flash attention, one wave per (b, h, 16-query tile).
// K chunk staged TRANSPOSED in LDS [dk][kv] (packed b32 stores),
// V chunk staged row-major via global_load_async_to_lds_b128 (ASYNCcnt),
// P bounced through LDS to re-layout into the A fragment,
// softmax row stats reduced with DPP16 (pure VALU).
// ---------------------------------------------------------------------------
__global__ void flash_attn(const unsigned short* __restrict__ Q,
                           const unsigned short* __restrict__ K,
                           const unsigned short* __restrict__ V,
                           unsigned short* __restrict__ Oc) {
    // per-wave LDS: K^T 64x32, V 32x64, P 16x32  (ushorts)
    __shared__ unsigned short lds[4 * (2048 + 2048 + 512)];

    const int lane = threadIdx.x & 31;
    const int wave = threadIdx.x >> 5;
    unsigned short* KlT = lds + wave * 4608;     // [dk][kv], row stride 32
    unsigned short* Vl  = KlT + 2048;            // [kv][dk], row stride 64
    unsigned short* Pl  = Vl + 2048;             // [m][k],  row stride 32

    const int tile = blockIdx.x * 4 + wave;      // 0..4095
    const int mt = tile & (SEQ / 16 - 1);
    const int h  = (tile >> 7) & (NH - 1);
    const int b  = tile >> 11;
    const int m0 = mt * 16;

    const int hi   = lane >> 4;
    const int l15  = lane & 15;
    const int kb   = l15 + 16 * hi;
    const int half = hi;                         // dk half for K transpose
    const float scale = 0.125f;                  // 1/sqrt(64)

    // Q A-fragments (two k=32 chunks covering DK=64), 2 x b128 each
    Frag16 aq[2];
    const unsigned short* qrow = Q + ((size_t)(b * SEQ + m0 + l15)) * DM + h * DK;
#pragma unroll
    for (int c = 0; c < 2; ++c) {
        aq[c].q2[0] = *reinterpret_cast<const v4u*>(qrow + c * 32 + 8 * hi);
        aq[c].q2[1] = *reinterpret_cast<const v4u*>(qrow + c * 32 + 16 + 8 * hi);
    }

    float mrow[8], lrow[8];
    v8f oacc[4] = {};
#pragma unroll
    for (int v = 0; v < 8; ++v) { mrow[v] = -1e30f; lrow[v] = 0.0f; }

    for (int j0 = 0; j0 <= m0 + 15; j0 += 32) {
        // ---- async V staging: lane copies kv row (j0+lane), 128B -> LDS ----
        {
            const unsigned short* vrow =
                V + ((size_t)(b * SEQ + j0 + lane)) * DM + h * DK;
            unsigned long long ga = (unsigned long long)vrow;
            unsigned ldsa = (unsigned)(unsigned long long)(Vl + (size_t)lane * 64);
            // previous iteration's LDS reads must be done before overwrite
            asm volatile("s_wait_dscnt 0" ::: "memory");
#pragma unroll
            for (int c = 0; c < 8; ++c) {
                asm volatile("global_load_async_to_lds_b128 %0, %1, off"
                             :: "v"(ldsa + c * 16),
                                "v"(ga + (unsigned long long)(c * 16))
                             : "memory");
            }
        }

        // ---- K staged transposed: lane owns kv pair (2q,2q+1), dk half ----
        {
            const int q2i = l15;                 // kv pair index
            const unsigned short* r0p =
                K + ((size_t)(b * SEQ + j0 + 2 * q2i)) * DM + h * DK + 32 * half;
            const unsigned short* r1p = r0p + DM;
            RowBuf ra, rb;
#pragma unroll
            for (int c = 0; c < 4; ++c) {
                ra.q[c] = reinterpret_cast<const v4u*>(r0p)[c];
                rb.q[c] = reinterpret_cast<const v4u*>(r1p)[c];
            }
#pragma unroll
            for (int dk = 0; dk < 32; ++dk) {
                unsigned pack = (unsigned)ra.u[dk] | ((unsigned)rb.u[dk] << 16);
                *reinterpret_cast<unsigned*>(
                    KlT + (size_t)(32 * half + dk) * 32 + 2 * q2i) = pack;
            }
        }

        // ---- scores: two 16x16 tiles, each 2 WMMAs over DK=64 ----
        v8f sc[2];
#pragma unroll
        for (int t = 0; t < 2; ++t) {
            v8f st = {};
#pragma unroll
            for (int c = 0; c < 2; ++c) {
                Frag16 bk;   // col (16t+l15) of K^T, dk rows c*32 + table-order k
#pragma unroll
                for (int s = 0; s < 16; ++s)
                    bk.u[s] = KlT[(size_t)(c * 32 + 8 * hi + s + (s & 8)) * 32 + 16 * t + l15];
                st = __builtin_amdgcn_wmma_f32_16x16x32_bf16(
                    false, aq[c].v, false, bk.v, (short)0, st, false, false);
            }
            sc[t] = st;
        }

        // ---- scale + causal mask + online softmax (DPP16 reductions) ----
#pragma unroll
        for (int v = 0; v < 8; ++v) {
            int mglob = m0 + v + 8 * hi;
            float s0 = sc[0][v] * scale;
            float s1 = sc[1][v] * scale;
            if (j0 + l15 > mglob)      s0 = -1e30f;
            if (j0 + 16 + l15 > mglob) s1 = -1e30f;

            float mx = row16_max(fmaxf(s0, s1));
            float mnew  = fmaxf(mrow[v], mx);
            float alpha = __expf(mrow[v] - mnew);
            float p0 = __expf(s0 - mnew);
            float p1 = __expf(s1 - mnew);
            float sum = row16_sum(p0 + p1);

            lrow[v] = lrow[v] * alpha + sum;
            mrow[v] = mnew;
#pragma unroll
            for (int t = 0; t < 4; ++t) oacc[t][v] *= alpha;

            Pl[(v + 8 * hi) * 32 + l15]      = f2bf(p0);
            Pl[(v + 8 * hi) * 32 + 16 + l15] = f2bf(p1);
        }

        // ---- P A-fragment: two contiguous 8-elem runs per lane ----
        Frag16 ap;
        ap.q2[0] = *reinterpret_cast<const v4u*>(Pl + l15 * 32 + 8 * hi);
        ap.q2[1] = *reinterpret_cast<const v4u*>(Pl + l15 * 32 + 16 + 8 * hi);

        // ---- V must have landed in LDS before PV fragment reads ----
        asm volatile("s_wait_asynccnt 0" ::: "memory");

        // ---- PV: 4 output tiles over DK=64, B = V column 16t+l15, kv in table order ----
#pragma unroll
        for (int t = 0; t < 4; ++t) {
            Frag16 bv;
#pragma unroll
            for (int s = 0; s < 16; ++s)
                bv.u[s] = Vl[(size_t)(8 * hi + s + (s & 8)) * 64 + 16 * t + l15];
            oacc[t] = __builtin_amdgcn_wmma_f32_16x16x32_bf16(
                false, ap.v, false, bv.v, (short)0, oacc[t], false, false);
        }
    }

    // ---- epilogue: normalize, store attn (concat layout B,S,H*DK) ----
#pragma unroll
    for (int t = 0; t < 4; ++t)
#pragma unroll
        for (int v = 0; v < 8; ++v) {
            int m = v + 8 * hi;
            float val = oacc[t][v] / lrow[v];
            Oc[((size_t)(b * SEQ + m0 + m)) * DM + h * DK + t * 16 + l15] = f2bf(val);
        }
}

// ---------------------------------------------------------------------------
// Host-side launcher
// ---------------------------------------------------------------------------
extern "C" void kernel_launch(void* const* d_in, const int* in_sizes, int n_in,
                              void* d_out, int out_size, void* d_ws, size_t ws_size,
                              hipStream_t stream) {
    const float* x   = (const float*)d_in[0];
    const int*   pos = (const int*)d_in[1];
    const float* Wq  = (const float*)d_in[2];
    const float* Wk  = (const float*)d_in[3];
    const float* Wv  = (const float*)d_in[4];
    const float* Wo  = (const float*)d_in[5];
    float* out = (float*)d_out;

    const size_t N_X = (size_t)BS * DM;   // 4,194,304
    const size_t N_W = (size_t)DM * DM;   // 1,048,576

    unsigned short* ws  = (unsigned short*)d_ws;
    unsigned short* xb  = ws;
    unsigned short* wqb = xb + N_X;
    unsigned short* wkb = wqb + N_W;
    unsigned short* wvb = wkb + N_W;
    unsigned short* wob = wvb + N_W;
    unsigned short* qb  = wob + N_W;
    unsigned short* kbuf= qb + N_X;
    unsigned short* vb  = kbuf + N_X;
    unsigned short* ab  = vb + N_X;

    // 1) fp32 -> bf16 (packed pairs)
    cvt_f32_bf16<<<(int)((N_X / 2 + 255) / 256), 256, 0, stream>>>(
        x, (unsigned int*)xb, (int)(N_X / 2));
    cvt_f32_bf16<<<(int)((N_W / 2 + 255) / 256), 256, 0, stream>>>(
        Wq, (unsigned int*)wqb, (int)(N_W / 2));
    cvt_f32_bf16<<<(int)((N_W / 2 + 255) / 256), 256, 0, stream>>>(
        Wk, (unsigned int*)wkb, (int)(N_W / 2));
    cvt_f32_bf16<<<(int)((N_W / 2 + 255) / 256), 256, 0, stream>>>(
        Wv, (unsigned int*)wvb, (int)(N_W / 2));
    cvt_f32_bf16<<<(int)((N_W / 2 + 255) / 256), 256, 0, stream>>>(
        Wo, (unsigned int*)wob, (int)(N_W / 2));

    // 2) Q/K/V projections: C = X * W^T  (M=4096, N=1024, K=1024)
    //    grid: (N/64, M/128), 256-thread blocks (8 waves)
    dim3 ggrid(DM / 64, BS / 128);
    gemm_xwT<false><<<ggrid, 256, 0, stream>>>(xb, wqb, qb, BS, DM, DM);
    gemm_xwT<false><<<ggrid, 256, 0, stream>>>(xb, wkb, kbuf, BS, DM, DM);
    gemm_xwT<false><<<ggrid, 256, 0, stream>>>(xb, wvb, vb, BS, DM, DM);

    // 3) RoPE on Q, K
    int pairs = BS * (DM / 2);
    rope_kernel<<<(pairs + 255) / 256, 256, 0, stream>>>(qb, kbuf, pos);

    // 4) flash attention: 4096 (b,h,m-tile) waves, 4 waves per block
    flash_attn<<<(BATCH * NH * (SEQ / 16)) / 4, 128, 0, stream>>>(qb, kbuf, vb, ab);

    // 5) output projection to fp32
    gemm_xwT<true><<<ggrid, 256, 0, stream>>>(ab, wob, out, BS, DM, DM);
}
